// ParticleNet_6588479832126
// MI455X (gfx1250) — matrix operand order repaired and checked by reading, not compiled, not run on hardware
//
#include <hip/hip_runtime.h>
#include <hip/hip_bf16.h>

#define NNODES  100000
#define NEDGES  1600000
#define NGRAPHS 256
#define CDIV(a, b) (((a) + (b) - 1) / (b))

typedef __attribute__((ext_vector_type(16))) __bf16 v16bf;
typedef __attribute__((ext_vector_type(8)))  __bf16 v8bf;
typedef __attribute__((ext_vector_type(4)))  __bf16 v4bf;
typedef __attribute__((ext_vector_type(8)))  float  v8f;

__device__ __forceinline__ __bf16 f2bf(float f) {
    union { float f; unsigned u; } v; v.f = f;
    unsigned r = v.u + 0x7FFFu + ((v.u >> 16) & 1u);   // round-to-nearest-even
    unsigned short h = (unsigned short)(r >> 16);
    return __builtin_bit_cast(__bf16, h);
}

// ---------------- elementwise helpers ----------------

__global__ void zero_f4_kernel(float4* __restrict__ p, int n4) {
    int t = blockIdx.x * blockDim.x + threadIdx.x;
    if (t < n4) p[t] = make_float4(0.f, 0.f, 0.f, 0.f);
}

__global__ void cvt_bf16_kernel(const float* __restrict__ in, __bf16* __restrict__ out, int n4) {
    int t = blockIdx.x * blockDim.x + threadIdx.x;
    if (t >= n4) return;
    float4 v = ((const float4*)in)[t];
    v4bf o;
    o[0] = f2bf(v.x); o[1] = f2bf(v.y); o[2] = f2bf(v.z); o[3] = f2bf(v.w);
    ((v4bf*)out)[t] = o;
}

// W is [K,N] row-major fp32 -> wt is [N,K] row-major bf16 (column-major B for WMMA)
__global__ void transpose_bf16_kernel(const float* __restrict__ W, __bf16* __restrict__ wt,
                                      int K, int N) {
    int t = blockIdx.x * blockDim.x + threadIdx.x;
    if (t >= K * N) return;
    int n = t / K, k = t % K;
    wt[t] = f2bf(W[(long)k * N + n]);
}

// ---------------- degree / normalization ----------------

__global__ void deg_count_kernel(const int* __restrict__ dst, float* __restrict__ deg, int E) {
    int t = blockIdx.x * blockDim.x + threadIdx.x;
    if (t < E) unsafeAtomicAdd(&deg[dst[t]], 1.0f);
}

__global__ void deg_finalize_kernel(const float* __restrict__ deg, float* __restrict__ dinv,
                                    float* __restrict__ selfn, int N) {
    int t = blockIdx.x * blockDim.x + threadIdx.x;
    if (t >= N) return;
    float d = deg[t] + 1.0f;           // self loop
    dinv[t]  = rsqrtf(d);
    selfn[t] = 1.0f / d;
}

// ---------------- WMMA bf16 GEMM: C[M,N] = A[M,K] @ B[K,N] ----------------
// A: bf16 row-major [M,K]; Bt: bf16 [N,K] (i.e. B column-major); C: fp32 [M,N].
// One 16x16 output tile per wave, 8 waves per 256-thread block, K stepped by 32.

__global__ void gemm_wmma_bf16_kernel(const __bf16* __restrict__ A, const __bf16* __restrict__ Bt,
                                      float* __restrict__ C, int M, int N, int K) {
    int wave = blockIdx.x * (blockDim.x >> 5) + (threadIdx.x >> 5);
    int ntn = N >> 4;
    int total = (M >> 4) * ntn;
    if (wave >= total) return;                 // wave-uniform branch: EXEC stays all-ones
    int tm = wave / ntn;
    int tn = wave % ntn;
    int lane = threadIdx.x & 31;
    int half = lane >> 4;
    int r    = lane & 15;

    // A 16-bit 16x32 layout: lanes 0-15 hold K={0..7,16..23}, lanes 16-31 hold K={8..15,24..31}
    const __bf16* arow = A  + (long)(tm * 16 + r) * K + half * 8;
    // B 16-bit 32x16 layout: lanes 0-15 hold K={0..7,16..23} of col r, lanes 16-31 hold K={8..15,24..31}
    const __bf16* brow = Bt + (long)(tn * 16 + r) * K + half * 8;

    v8f acc = {};
    for (int k0 = 0; k0 < K; k0 += 32) {
        v8bf alo = *(const v8bf*)(arow + k0);
        v8bf ahi = *(const v8bf*)(arow + k0 + 16);
        v8bf blo = *(const v8bf*)(brow + k0);
        v8bf bhi = *(const v8bf*)(brow + k0 + 16);
        __builtin_prefetch(arow + k0 + 32, 0, 1);
        v16bf a, b;
#pragma unroll
        for (int i = 0; i < 8; ++i) {
            a[i] = alo[i]; a[i + 8] = ahi[i];
            b[i] = blo[i]; b[i + 8] = bhi[i];
        }
        acc = __builtin_amdgcn_wmma_f32_16x16x32_bf16(false, a, false, b,
                                                      (short)0, acc, false, false);
    }

    int col = tn * 16 + r;
#pragma unroll
    for (int j = 0; j < 8; ++j) {
        int row = tm * 16 + half * 8 + j;      // C: VGPR j -> M = 8*half + j, lane -> N
        C[(long)row * N + col] = acc[j];
    }
}

// ---------------- edge scatter-add (the HBM-bound hot loop) ----------------

__global__ void edge_agg_kernel(const float* __restrict__ h, const int* __restrict__ src,
                                const int* __restrict__ dst, const float* __restrict__ dinv,
                                float* __restrict__ agg, int E, int d4shift) {
    int t = blockIdx.x * blockDim.x + threadIdx.x;
    int e = t >> d4shift;
    if (e >= E) return;
    int c  = t & ((1 << d4shift) - 1);
    int d4 = 1 << d4shift;
    int s = src[e], d = dst[e];
    float nrm = dinv[s] * dinv[d];
    float4 hv = ((const float4*)h)[(long)s * d4 + c];
    float* ag = agg + ((long)d * d4 + c) * 4;
    unsafeAtomicAdd(ag + 0, hv.x * nrm);       // native global_atomic_add_f32
    unsafeAtomicAdd(ag + 1, hv.y * nrm);
    unsafeAtomicAdd(ag + 2, hv.z * nrm);
    unsafeAtomicAdd(ag + 3, hv.w * nrm);
}

// out = relu(agg + h*self_norm + b) (+ residual)
__global__ void combine_kernel(const float* __restrict__ agg, const float* __restrict__ h,
                               const float* __restrict__ selfn, const float* __restrict__ bias,
                               const float* __restrict__ xini, float* __restrict__ out,
                               int total, int dshift) {
    int t = blockIdx.x * blockDim.x + threadIdx.x;
    if (t >= total) return;
    int n = t >> dshift;
    int f = t & ((1 << dshift) - 1);
    float v = agg[t] + h[t] * selfn[n] + bias[f];
    v = fmaxf(v, 0.f);
    if (xini) v += xini[t];
    out[t] = v;
}

// ---------------- global mean pool (dim = 256) ----------------

__global__ void pool_accum_kernel(const float* __restrict__ x, const int* __restrict__ batch,
                                  float* __restrict__ gsum, float* __restrict__ gcnt) {
    int t = blockIdx.x * blockDim.x + threadIdx.x;
    if (t >= NNODES * 64) return;              // 64 float4 chunks per node
    int n = t >> 6, c = t & 63;
    int g = batch[n];
    float4 v = ((const float4*)x)[t];
    float* gs = gsum + (long)g * 256 + c * 4;
    unsafeAtomicAdd(gs + 0, v.x);
    unsafeAtomicAdd(gs + 1, v.y);
    unsafeAtomicAdd(gs + 2, v.z);
    unsafeAtomicAdd(gs + 3, v.w);
    if (c == 0) unsafeAtomicAdd(&gcnt[g], 1.0f);
}

__global__ void pool_final_kernel(const float* __restrict__ gsum, const float* __restrict__ gcnt,
                                  __bf16* __restrict__ gbf) {
    int t = blockIdx.x * blockDim.x + threadIdx.x;
    if (t >= NGRAPHS * 256) return;
    int g = t >> 8;
    float cnt = fmaxf(gcnt[g], 1.0f);
    gbf[t] = f2bf(gsum[t] / cnt);
}

__global__ void bias_relu_kernel(float* __restrict__ h, const float* __restrict__ bias, int total) {
    int t = blockIdx.x * blockDim.x + threadIdx.x;
    if (t >= total) return;
    int f = t & 511;                           // cols = 512
    h[t] = fmaxf(h[t] + bias[f], 0.f);
}

// 512 -> 2 linear + relu + log_softmax; one wave per graph
__global__ void mlp_final_kernel(const float* __restrict__ h, const float* __restrict__ Wm1,
                                 const float* __restrict__ bm1, float* __restrict__ out) {
    int wid = blockIdx.x * (blockDim.x >> 5) + (threadIdx.x >> 5);
    if (wid >= NGRAPHS) return;
    int lane = threadIdx.x & 31;
    float p0 = 0.f, p1 = 0.f;
    for (int k = lane; k < 512; k += 32) {
        float v = h[(long)wid * 512 + k];
        p0 += v * Wm1[k * 2 + 0];
        p1 += v * Wm1[k * 2 + 1];
    }
#pragma unroll
    for (int o = 16; o > 0; o >>= 1) {
        p0 += __shfl_xor(p0, o, 32);
        p1 += __shfl_xor(p1, o, 32);
    }
    if (lane == 0) {
        float z0 = fmaxf(p0 + bm1[0], 0.f);
        float z1 = fmaxf(p1 + bm1[1], 0.f);
        float m  = fmaxf(z0, z1);
        float l  = m + logf(expf(z0 - m) + expf(z1 - m));
        out[wid * 2 + 0] = z0 - l;
        out[wid * 2 + 1] = z1 - l;
    }
}

// ---------------- host orchestration ----------------

extern "C" void kernel_launch(void* const* d_in, const int* in_sizes, int n_in,
                              void* d_out, int out_size, void* d_ws, size_t ws_size,
                              hipStream_t stream) {
    (void)in_sizes; (void)n_in; (void)out_size; (void)ws_size;

    const float* x     = (const float*)d_in[0];
    const int*   ei    = (const int*)d_in[1];
    const int*   src   = ei;
    const int*   dst   = ei + NEDGES;
    const int*   batch = (const int*)d_in[2];
    const float* W[9]; const float* B[9];
    for (int i = 0; i < 9; ++i) {
        W[i] = (const float*)d_in[3 + 2 * i];
        B[i] = (const float*)d_in[4 + 2 * i];
    }
    const float* Wm0 = (const float*)d_in[21];
    const float* bm0 = (const float*)d_in[22];
    const float* Wm1 = (const float*)d_in[23];
    const float* bm1 = (const float*)d_in[24];

    const int dims[10] = {64, 64, 32, 64, 128, 64, 128, 256, 128, 256};

    char* ws = (char*)d_ws;
    size_t off = 0;
    auto take = [&](size_t bytes) -> char* {
        char* p = ws + off;
        off = (off + bytes + 255) & ~(size_t)255;
        return p;
    };

    float*  act0  = (float*)take((size_t)NNODES * 256 * 4);
    float*  act1  = (float*)take((size_t)NNODES * 256 * 4);
    float*  hbuf  = (float*)take((size_t)NNODES * 256 * 4);
    float*  aggb  = (float*)take((size_t)NNODES * 256 * 4);
    __bf16* xbf   = (__bf16*)take((size_t)NNODES * 256 * 2);
    float*  dinv  = (float*)take((size_t)NNODES * 4);
    float*  selfn = (float*)take((size_t)NNODES * 4);
    __bf16* wt[10];
    for (int i = 0; i < 9; ++i) wt[i] = (__bf16*)take((size_t)dims[i] * dims[i + 1] * 2);
    wt[9] = (__bf16*)take((size_t)256 * 512 * 2);
    float*  gsum = (float*)take((size_t)NGRAPHS * 256 * 4);
    float*  gcnt = (float*)take((size_t)NGRAPHS * 4);
    __bf16* gbf  = (__bf16*)take((size_t)NGRAPHS * 256 * 2);
    float*  mlph = (float*)take((size_t)NGRAPHS * 512 * 4);

    // --- weights: transpose + bf16 convert (cheap, once per call) ---
    for (int i = 0; i < 9; ++i) {
        int kn = dims[i] * dims[i + 1];
        transpose_bf16_kernel<<<CDIV(kn, 256), 256, 0, stream>>>(W[i], wt[i], dims[i], dims[i + 1]);
    }
    transpose_bf16_kernel<<<CDIV(256 * 512, 256), 256, 0, stream>>>(Wm0, wt[9], 256, 512);

    // --- degrees / GCN normalization (deg scratch = aggb) ---
    zero_f4_kernel<<<CDIV(NNODES / 4, 256), 256, 0, stream>>>((float4*)aggb, NNODES / 4);
    deg_count_kernel<<<CDIV(NEDGES, 256), 256, 0, stream>>>(dst, aggb, NEDGES);
    deg_finalize_kernel<<<CDIV(NNODES, 256), 256, 0, stream>>>(aggb, dinv, selfn, NNODES);

    auto gemm = [&](const __bf16* A, const __bf16* Bt, float* C, int M, int Nd, int K) {
        int tiles = (M >> 4) * (Nd >> 4);
        gemm_wmma_bf16_kernel<<<CDIV(tiles, 8), 256, 0, stream>>>(A, Bt, C, M, Nd, K);
    };

    // --- 9 GCN layers ---
    const float* cur = x;
    float* acts[2] = {act0, act1};
    for (int i = 0; i < 9; ++i) {
        int K = dims[i], Nd = dims[i + 1];
        cvt_bf16_kernel<<<CDIV(NNODES * K / 4, 256), 256, 0, stream>>>(cur, xbf, NNODES * K / 4);
        gemm(xbf, wt[i], hbuf, NNODES, Nd, K);
        zero_f4_kernel<<<CDIV(NNODES * Nd / 4, 256), 256, 0, stream>>>((float4*)aggb, NNODES * Nd / 4);
        int d4s = __builtin_ctz(Nd / 4);
        edge_agg_kernel<<<CDIV(NEDGES * (Nd / 4), 256), 256, 0, stream>>>(
            hbuf, src, dst, dinv, aggb, NEDGES, d4s);
        int dsh = __builtin_ctz(Nd);
        const float* xini = (i == 2) ? x : nullptr;   // residual from x in layer 2 only (its 64-wide output matches x)
        float* outp = acts[i & 1];
        combine_kernel<<<CDIV(NNODES * Nd, 256), 256, 0, stream>>>(
            aggb, hbuf, selfn, B[i], xini, outp, NNODES * Nd, dsh);
        cur = outp;
    }

    // --- global mean pool -> bf16 [256,256] ---
    zero_f4_kernel<<<CDIV(NGRAPHS * 256 / 4, 256), 256, 0, stream>>>((float4*)gsum, NGRAPHS * 256 / 4);
    zero_f4_kernel<<<1, 64, 0, stream>>>((float4*)gcnt, NGRAPHS / 4);
    pool_accum_kernel<<<CDIV(NNODES * 64, 256), 256, 0, stream>>>(cur, batch, gsum, gcnt);
    pool_final_kernel<<<CDIV(NGRAPHS * 256, 256), 256, 0, stream>>>(gsum, gcnt, gbf);

    // --- MLP head: [256,256]@[256,512] WMMA, bias+relu, 512->2 + log_softmax ---
    gemm(gbf, wt[9], mlph, NGRAPHS, 512, 256);
    bias_relu_kernel<<<CDIV(NGRAPHS * 512, 256), 256, 0, stream>>>(mlph, bm0, NGRAPHS * 512);
    mlp_final_kernel<<<CDIV(NGRAPHS, 8), 256, 0, stream>>>(mlph, Wm1, bm1, (float*)d_out);
}
